// GraphQueryAttention_63118839382029
// MI455X (gfx1250) — matrix operand order repaired and checked by reading, not compiled, not run on hardware
//
#include <hip/hip_runtime.h>

typedef __attribute__((ext_vector_type(16))) __bf16 v16bf;
typedef __attribute__((ext_vector_type(8)))  __bf16 v8bf;
typedef __attribute__((ext_vector_type(8)))  float  v8f;
typedef __attribute__((ext_vector_type(4)))  unsigned int u32x4;
typedef __attribute__((ext_vector_type(8)))  int i32x8;
typedef __attribute__((ext_vector_type(4)))  int i32x4;

constexpr int B_   = 2;
constexpr int S_   = 2048;
constexpr int D_   = 4096;
constexpr int H_   = 32;
constexpr int KVH_ = 8;
constexpr int HD_  = D_ / H_;        // 128
constexpr int NREP_ = H_ / KVH_;     // 4
constexpr float SCALE_ = 0.08838834764831845f;  // 1/sqrt(128)

// ---------------------------------------------------------------------------
// fp32 -> bf16 conversion
// ---------------------------------------------------------------------------
__global__ __launch_bounds__(256)
void f32_to_bf16_kernel(const float* __restrict__ in, __bf16* __restrict__ out, long n) {
    long i = (long)blockIdx.x * blockDim.x + threadIdx.x;
    if (i < n) out[i] = (__bf16)in[i];
}

// ---------------------------------------------------------------------------
// TDM: issue a 128x64 bf16 tile load (global -> LDS) via the Tensor Data
// Mover. Descriptor per CDNA5 ISA 8.3/8.4:
//   group0: count=1 | lds_addr | global_addr(57b) | type=2
//   group1: data_size=1(2B), tensor_dim0=64, tensor_dim1=128,
//           tile_dim0=64, tile_dim1=128, tensor_dim0_stride=row stride
// Issued once per wave (TDM ignores EXEC); tracked with TENSORcnt.
// This toolchain uses the 6-arg builtin (clang-23 / therock-10.0 headers).
// ---------------------------------------------------------------------------
__device__ __forceinline__ void tdm_load_tile_128x64(unsigned lds_byte_off,
                                                     const __bf16* gptr,
                                                     int row_stride_elems) {
    unsigned long long ga = (unsigned long long)gptr;
    u32x4 g0;
    g0[0] = 1u;                                           // count=1, user mode
    g0[1] = lds_byte_off;                                 // LDS byte address
    g0[2] = (unsigned)(ga & 0xFFFFFFFFu);                 // global addr [31:0]
    g0[3] = (unsigned)((ga >> 32) & 0x01FFFFFFu) | (2u << 30);  // [56:32] | type=2
    i32x8 g1;
    g1[0] = 0x00010000;                  // mask=0, data_size=1 (2 bytes), no pad
    g1[1] = (int)(64u << 16);            // tensor_dim0[15:0]=64 in bits 63:48
    g1[2] = (int)(128u << 16);           // tensor_dim0[31:16]=0 | tensor_dim1[15:0]=128
    g1[3] = (int)(64u << 16);            // tensor_dim1[31:16]=0 | tile_dim0=64
    g1[4] = 128;                         // tile_dim1=128 | tile_dim2=0
    g1[5] = row_stride_elems;            // tensor_dim0_stride[31:0]
    g1[6] = 0;                           // stride0[47:32]=0 | stride1[15:0]=0
    g1[7] = 0;
    i32x4 z4 = {0, 0, 0, 0};
    i32x8 z8 = {0, 0, 0, 0, 0, 0, 0, 0};
    __builtin_amdgcn_tensor_load_to_lds(g0, g1, z4, z4, z8, 0);
}

// ---------------------------------------------------------------------------
// NT GEMM:  C[M,N] = A[M,K] * W[N,K]^T
// 256 threads (8 waves) per 128x128 block tile, K-panel = 64, LDS
// double-buffered, panels DMA'd by the TDM (wave 0 issues, s_wait_tensorcnt
// gates). Each wave owns a 32x64 register tile (2x4 WMMA accumulators).
// Fragment layouts (bf16, 16x16x32):
//   A: lane L -> row M=L%16; half i -> K = 16*(L>=16)+i
//   B: lane L -> col N=L%16; half i -> K = 16*(L>=16)+i
//   C: VGPR r, lane L -> M = r+8*(L>=16), N = L%16
// ---------------------------------------------------------------------------
constexpr int BM = 128, BN = 128, BK = 64;

template <bool F32OUT>
__global__ __launch_bounds__(256)
void gemm_nt_wmma(const __bf16* __restrict__ A, const __bf16* __restrict__ W,
                  float* __restrict__ Cf, __bf16* __restrict__ Cb,
                  int M, int N, int K) {
    __shared__ __bf16 Alds[2][BM * BK];   // 16KB each
    __shared__ __bf16 Blds[2][BN * BK];   // 16KB each

    const int nbn  = N / BN;
    const int tm   = (blockIdx.x / nbn) * BM;
    const int tn   = (blockIdx.x % nbn) * BN;
    const int lane = threadIdx.x & 31;
    const int wave = __builtin_amdgcn_readfirstlane((int)(threadIdx.x >> 5));
    const int half = lane >> 4;
    const int l16  = lane & 15;
    const int wm   = (wave >> 1) * 32;    // wave M offset within block (0..96)
    const int wn   = (wave & 1) * 64;     // wave N offset within block (0,64)

    const __bf16* Ab = A + (long)tm * K;
    const __bf16* Wb = W + (long)tn * K;

    v8f acc[2][4] = {};

    const int nk = K / BK;
    if (wave == 0) {
        tdm_load_tile_128x64((unsigned)(unsigned long long)&Alds[0][0], Ab, K);
        tdm_load_tile_128x64((unsigned)(unsigned long long)&Blds[0][0], Wb, K);
    }

    for (int kt = 0; kt < nk; ++kt) {
        const int buf = kt & 1;
        const bool more = (kt + 1) < nk;
        if (wave == 0) {
            if (more) {
                const long knext = (long)(kt + 1) * BK;
                tdm_load_tile_128x64((unsigned)(unsigned long long)&Alds[buf ^ 1][0],
                                     Ab + knext, K);
                tdm_load_tile_128x64((unsigned)(unsigned long long)&Blds[buf ^ 1][0],
                                     Wb + knext, K);
                __builtin_amdgcn_s_wait_tensorcnt((short)2);  // older pair done
            } else {
                __builtin_amdgcn_s_wait_tensorcnt((short)0);
            }
        }
        __syncthreads();

        const __bf16* As = &Alds[buf][0];
        const __bf16* Bs = &Blds[buf][0];
#pragma unroll
        for (int kk = 0; kk < BK; kk += 32) {
            v16bf af[2];
#pragma unroll
            for (int mi = 0; mi < 2; ++mi) {
                const int row = wm + mi * 16 + l16;
                const int ab  = kk + (half ? 16 : 0);
                v8bf lo = *(const v8bf*)(As + row * BK + ab);
                v8bf hi = *(const v8bf*)(As + row * BK + ab + 8);
#pragma unroll
                for (int i = 0; i < 8; ++i) { af[mi][i] = lo[i]; af[mi][i + 8] = hi[i]; }
            }
            v16bf bfv[4];
#pragma unroll
            for (int ni = 0; ni < 4; ++ni) {
                const int n = wn + ni * 16 + l16;
                bfv[ni] = *(const v16bf*)(Bs + n * BK + kk + (half ? 16 : 0));
            }
#pragma unroll
            for (int mi = 0; mi < 2; ++mi)
#pragma unroll
                for (int ni = 0; ni < 4; ++ni)
                    acc[mi][ni] = __builtin_amdgcn_wmma_f32_16x16x32_bf16(
                        false, af[mi], false, bfv[ni], (short)0, acc[mi][ni],
                        false, false);
        }
        __syncthreads();
    }

#pragma unroll
    for (int mi = 0; mi < 2; ++mi)
#pragma unroll
        for (int ni = 0; ni < 4; ++ni)
#pragma unroll
            for (int r = 0; r < 8; ++r) {
                const long row = tm + wm + mi * 16 + half * 8 + r;
                const long idx = row * (long)N + tn + wn + ni * 16 + l16;
                if constexpr (F32OUT) Cf[idx] = acc[mi][ni][r];
                else                  Cb[idx] = (__bf16)acc[mi][ni][r];
            }
}

// ---------------------------------------------------------------------------
// Flash attention: one wave per (b, h, 16 query rows).
// Streams over keys in chunks of 32 with online softmax.
// ---------------------------------------------------------------------------
constexpr int VPAD = 8;  // pad LDS V rows to avoid bank conflicts

__global__ __launch_bounds__(32)
void attn_wmma(const __bf16* __restrict__ Q, const __bf16* __restrict__ Km,
               const __bf16* __restrict__ Vm, const float* __restrict__ bias,
               const int* __restrict__ mask, __bf16* __restrict__ Ho) {
    const int qt   = blockIdx.x;          // query tile (16 rows)
    const int h    = blockIdx.y;
    const int b    = blockIdx.z;
    const int kvh  = h / NREP_;
    const int lane = threadIdx.x;
    const int half = lane >> 4;
    const int l16  = lane & 15;

    __shared__ __bf16 vlds[32 * (HD_ + VPAD)];  // 32 x 128 V tile (padded)
    __shared__ __bf16 plds[16 * 32];            // 16 x 32 probability tile

    const __bf16* qrow = Q + ((long)(b * S_ + qt * 16 + l16) * H_ + h) * HD_;

    float m_i[8], l_i[8], alpha[8];
    v8f o[8] = {};
#pragma unroll
    for (int r = 0; r < 8; ++r) { m_i[r] = -3.0e38f; l_i[r] = 0.0f; }

    for (int j = 0; j < S_; j += 32) {
        // prefetch next bias chunk (speculative; dropped if OOB)
        __builtin_prefetch(bias + ((long)h * S_ + qt * 16 + half * 8) * S_ + j + 128, 0, 0);

        // ---- stage V rows j..j+31 into LDS (each lane copies one row) ----
        const __bf16* vrow = Vm + ((long)(b * S_ + j + lane) * KVH_ + kvh) * HD_;
#pragma unroll
        for (int t = 0; t < HD_; t += 8)
            *(v8bf*)&vlds[lane * (HD_ + VPAD) + t] = *(const v8bf*)(vrow + t);

        // ---- scores: two 16x16 tiles (keys j..j+15 and j+16..j+31) ----
        v8f s0 = {}, s1 = {};
        const __bf16* krow0 = Km + ((long)(b * S_ + j + l16) * KVH_ + kvh) * HD_;
        const __bf16* krow1 = krow0 + (long)16 * KVH_ * HD_;
#pragma unroll
        for (int d0 = 0; d0 < HD_; d0 += 32) {
            const int ab = d0 + (half ? 16 : 0);
            v8bf qlo = *(const v8bf*)(qrow + ab);
            v8bf qhi = *(const v8bf*)(qrow + ab + 8);
            v16bf qf;
#pragma unroll
            for (int i = 0; i < 8; ++i) { qf[i] = qlo[i]; qf[i + 8] = qhi[i]; }
            v16bf k0f = *(const v16bf*)(krow0 + d0 + (half ? 16 : 0));
            s0 = __builtin_amdgcn_wmma_f32_16x16x32_bf16(
                false, qf, false, k0f, (short)0, s0, false, false);
            v16bf k1f = *(const v16bf*)(krow1 + d0 + (half ? 16 : 0));
            s1 = __builtin_amdgcn_wmma_f32_16x16x32_bf16(
                false, qf, false, k1f, (short)0, s1, false, false);
        }

        // ---- scale + bias + mask + online softmax (row = r + 8*half) ----
        const int mk0 = mask[b * S_ + j + l16];
        const int mk1 = mask[b * S_ + j + 16 + l16];
#pragma unroll
        for (int r = 0; r < 8; ++r) {
            const long rowg = qt * 16 + half * 8 + r;
            const float bi0 = bias[((long)h * S_ + rowg) * S_ + j + l16];
            const float bi1 = bias[((long)h * S_ + rowg) * S_ + j + 16 + l16];
            float sc0 = s0[r] * SCALE_ + bi0;
            float sc1 = s1[r] * SCALE_ + bi1;
            if (mk0 == 0) sc0 = -3.0e38f;
            if (mk1 == 0) sc1 = -3.0e38f;

            float mx = fmaxf(sc0, sc1);
#pragma unroll
            for (int off = 1; off < 16; off <<= 1)
                mx = fmaxf(mx, __shfl_xor(mx, off, 32));
            const float mnew = fmaxf(m_i[r], mx);
            alpha[r] = __expf(m_i[r] - mnew);
            m_i[r]   = mnew;

            const float p0 = __expf(sc0 - mnew);
            const float p1 = __expf(sc1 - mnew);
            float rs = p0 + p1;
#pragma unroll
            for (int off = 1; off < 16; off <<= 1)
                rs += __shfl_xor(rs, off, 32);
            l_i[r] = l_i[r] * alpha[r] + rs;

            plds[(half * 8 + r) * 32 + l16]      = (__bf16)p0;
            plds[(half * 8 + r) * 32 + 16 + l16] = (__bf16)p1;
        }
        __syncthreads();

        // ---- rescale running output ----
#pragma unroll
        for (int dt = 0; dt < 8; ++dt)
#pragma unroll
            for (int r = 0; r < 8; ++r) o[dt][r] *= alpha[r];

        // ---- P fragment (A layout, 16x32 over the key chunk) ----
        const int pb = half ? 16 : 0;
        v8bf pl = *(const v8bf*)&plds[l16 * 32 + pb];
        v8bf ph = *(const v8bf*)&plds[l16 * 32 + pb + 8];
        v16bf pf;
#pragma unroll
        for (int i = 0; i < 8; ++i) { pf[i] = pl[i]; pf[i + 8] = ph[i]; }

        // ---- P @ V: 8 WMMAs over HD=128 output columns ----
        const int kb = half * 16;
#pragma unroll
        for (int dt = 0; dt < 8; ++dt) {
            v16bf vf;
#pragma unroll
            for (int i = 0; i < 16; ++i)
                vf[i] = vlds[(kb + i) * (HD_ + VPAD) + dt * 16 + l16];
            o[dt] = __builtin_amdgcn_wmma_f32_16x16x32_bf16(
                false, pf, false, vf, (short)0, o[dt], false, false);
        }
        __syncthreads();
    }

    // ---- normalize and store head output (bf16, [B,S,H,HD] row-major) ----
#pragma unroll
    for (int r = 0; r < 8; ++r) {
        const float inv  = 1.0f / l_i[r];
        const long  srow = (long)b * S_ + qt * 16 + half * 8 + r;
#pragma unroll
        for (int dt = 0; dt < 8; ++dt)
            Ho[(srow * H_ + h) * HD_ + dt * 16 + l16] = (__bf16)(o[dt][r] * inv);
    }
}

// ---------------------------------------------------------------------------
// Host-side orchestration
// ---------------------------------------------------------------------------
extern "C" void kernel_launch(void* const* d_in, const int* in_sizes, int n_in,
                              void* d_out, int out_size, void* d_ws, size_t ws_size,
                              hipStream_t stream) {
    (void)in_sizes; (void)n_in; (void)out_size; (void)ws_size;

    const float* X    = (const float*)d_in[0];
    const int*   msk  = (const int*)d_in[1];
    const float* bias = (const float*)d_in[2];
    const float* Wq   = (const float*)d_in[3];
    const float* Wk   = (const float*)d_in[4];
    const float* Wv   = (const float*)d_in[5];
    const float* Wo   = (const float*)d_in[6];
    float*       out  = (float*)d_out;

    char*  ws  = (char*)d_ws;
    size_t off = 0;
    auto alloc = [&](size_t bytes) -> char* {
        char* p = ws + off;
        off += (bytes + 255) & ~(size_t)255;
        return p;
    };

    const long nX  = (long)B_ * S_ * D_;          // 16.8M
    const long nWq = (long)D_ * D_;               // 16.8M
    const long nWk = (long)KVH_ * HD_ * D_;       // 4.2M
    const long nKV = (long)B_ * S_ * KVH_ * HD_;  // 4.2M

    __bf16* Xb  = (__bf16*)alloc(nX * 2);
    __bf16* Wqb = (__bf16*)alloc(nWq * 2);
    __bf16* Wkb = (__bf16*)alloc(nWk * 2);
    __bf16* Wvb = (__bf16*)alloc(nWk * 2);
    __bf16* Wob = (__bf16*)alloc(nWq * 2);
    __bf16* Qb  = (__bf16*)alloc(nX * 2);
    __bf16* Kb  = (__bf16*)alloc(nKV * 2);
    __bf16* Vb  = (__bf16*)alloc(nKV * 2);
    __bf16* Hb  = Xb;  // head_out reuses X's slot (X dead after projections)

    auto cvt = [&](const float* src, __bf16* dst, long n) {
        f32_to_bf16_kernel<<<dim3((unsigned)((n + 255) / 256)), dim3(256), 0, stream>>>(src, dst, n);
    };
    cvt(X,  Xb,  nX);
    cvt(Wq, Wqb, nWq);
    cvt(Wk, Wkb, nWk);
    cvt(Wv, Wvb, nWk);
    cvt(Wo, Wob, nWq);

    const int M = B_ * S_;  // 4096
    gemm_nt_wmma<false><<<dim3((M / BM) * (D_ / BN)), dim3(256), 0, stream>>>(
        Xb, Wqb, nullptr, Qb, M, D_, D_);
    gemm_nt_wmma<false><<<dim3((M / BM) * ((KVH_ * HD_) / BN)), dim3(256), 0, stream>>>(
        Xb, Wkb, nullptr, Kb, M, KVH_ * HD_, D_);
    gemm_nt_wmma<false><<<dim3((M / BM) * ((KVH_ * HD_) / BN)), dim3(256), 0, stream>>>(
        Xb, Wvb, nullptr, Vb, M, KVH_ * HD_, D_);

    attn_wmma<<<dim3(S_ / 16, H_, B_), dim3(32), 0, stream>>>(
        Qb, Kb, Vb, bias, msk, Hb);

    gemm_nt_wmma<true><<<dim3((M / BM) * (D_ / BN)), dim3(256), 0, stream>>>(
        Hb, Wob, out, nullptr, M, D_, D_);
}
